// SecondOrderInteraction_20358144983117
// MI455X (gfx1250) — compile-verified, matrix operand order checked by reading, not run on hardware
//
#include <hip/hip_runtime.h>

// FM second-order interaction on MI455X (gfx1250), wave32 + bf16 WMMA.
// res[b] = 0.5 * sum_k ((x@v)_k^2 - (x^2 @ v^2)_k)
//
// Roofline: x stream = 128 MB @ 23.3 TB/s -> ~5.5us floor; bf16 WMMA compute
// ~1.7us. B fragments (v / v^2, pre-swizzled bf16) staged into LDS with
// global_load_async_to_lds_b128 (double buffered), shared by 8 waves/block.
// x^2@v^2 is linear in F -> reduced over K in-kernel (shuffle butterfly);
// only x@v per-k partials (squared after full-F sum) go through scratch.

typedef __attribute__((ext_vector_type(16))) __bf16 v16bf;
typedef __attribute__((ext_vector_type(8)))  float  v8f;
typedef __attribute__((ext_vector_type(4)))  float  v4f;
typedef __attribute__((ext_vector_type(4)))  int    v4i;

#define B_ROWS 8192
#define F_DIM  4096
#define K_DIM  64
#define FSPLIT 4            // F-range split: 4 blocks cover one slab-group's F
#define WAVES_PER_BLOCK 8   // 256 threads, 8 wave32, 8 x 16 = 128 rows/block
#define FTILES (F_DIM / 32)
#define ITERS  (FTILES / FSPLIT)
#define TILE_BYTES 8192     // per f-tile: vb frag 4KB + v2b frag 4KB

// ---- async global->LDS copy + wait (builtin if present, else CDNA5 asm) ----
__device__ __forceinline__ void async_copy_b128(const void* g, void* l) {
#if __has_builtin(__builtin_amdgcn_global_load_async_to_lds_b128)
  typedef __attribute__((address_space(1))) v4i* gp_t;
  typedef __attribute__((address_space(3))) v4i* lp_t;
  __builtin_amdgcn_global_load_async_to_lds_b128((gp_t)(g), (lp_t)(l), 0, 0);
#else
  typedef __attribute__((address_space(3))) void* lv_t;
  unsigned l32 = (unsigned)(size_t)(lv_t)(l);
  asm volatile("global_load_async_to_lds_b128 %0, %1, off"
               :: "v"(l32), "v"((unsigned long long)(size_t)g) : "memory");
#endif
}

__device__ __forceinline__ void wait_async0() {
#if __has_builtin(__builtin_amdgcn_s_wait_asynccnt)
  __builtin_amdgcn_s_wait_asynccnt(0);
#else
  asm volatile("s_wait_asynccnt 0x0" ::: "memory");
#endif
}

// ---------------------------------------------------------------------------
// Prep: convert v (fp32 [F,K]) into bf16 v / v^2, pre-swizzled into WMMA
// B-matrix (32x16, 16-bit) per-lane fragment order, tiles interleaved:
// tile t (8KB): [vb frags: (nt*32+lane)*16 elems][v2b frags at +2048 elems].
// B layout (ISA 7.12.2, 16-bit): lane L holds column n=L&15; half=L>>4;
// elems 0..7 = K(local) = half*8+{0..7}; elems 8..15 = 16 + half*8 + {0..7}.
// ---------------------------------------------------------------------------
__global__ void fm_prep_v(const float* __restrict__ v, __bf16* __restrict__ vcomb) {
  int tid  = blockIdx.x * blockDim.x + threadIdx.x; // one per (ftile, ntile, lane)
  int lane = tid & 31;
  int nt   = (tid >> 5) & 3;
  int t    = tid >> 7;
  if (t >= FTILES) return;
  int n    = lane & 15;
  int half = lane >> 4;
  int col  = nt * 16 + n;

  v16bf a, a2;
#pragma unroll
  for (int j = 0; j < 8; ++j) {
    int f0 = t * 32 + half * 8 + j;
    float x0 = v[(size_t)f0 * K_DIM + col];
    a[j]      = (__bf16)x0;
    a2[j]     = (__bf16)(x0 * x0);
    int f1 = t * 32 + 16 + half * 8 + j;
    float x1 = v[(size_t)f1 * K_DIM + col];
    a[8 + j]  = (__bf16)x1;
    a2[8 + j] = (__bf16)(x1 * x1);
  }
  v16bf* base = (v16bf*)(vcomb + (size_t)t * 4096);
  base[nt * 32 + lane]       = a;   // vb   fragment region [0, 4KB)
  base[128 + nt * 32 + lane] = a2;  // v2b  fragment region [4KB, 8KB)
}

// ---------------------------------------------------------------------------
// Main: block = 8 waves sharing one F-split s; wave w owns a 16-row slab.
// Per f-tile, the 8KB B tile is async-copied into LDS (double-buffered);
// all 8 waves preload the 8 B fragments and run the 8-WMMA chain.
// ---------------------------------------------------------------------------
__global__ void __launch_bounds__(WAVES_PER_BLOCK * 32)
fm_main(const float* __restrict__ x, const __bf16* __restrict__ vcomb,
        float* __restrict__ c1p, float* __restrict__ c2p) {
  __shared__ __align__(16) char ldsb[2][TILE_BYTES];

  const int lane = threadIdx.x & 31;
  const int wave = threadIdx.x >> 5;
  const int s         = blockIdx.x % FSPLIT;   // shared F-split for the block
  const int slabgroup = blockIdx.x / FSPLIT;
  const int slab = slabgroup * WAVES_PER_BLOCK + wave;
  const int rowbase = slab * 16;
  const int m    = lane & 15;
  const int half = lane >> 4;
  const int kb   = half * 8;

  v8f c1[4] = {};
  v8f c2[4] = {};

  const float* xrow = x + (size_t)(rowbase + m) * F_DIM;
  const char* vtiles = (const char*)vcomb + (size_t)s * ITERS * TILE_BYTES;
  const int cidx = (int)threadIdx.x * 32;      // 256 lanes x 32B = 8KB tile

  // prologue: stage tile 0 into buffer 0
  async_copy_b128(vtiles + cidx,      &ldsb[0][cidx]);
  async_copy_b128(vtiles + cidx + 16, &ldsb[0][cidx + 16]);

  for (int it = 0; it < ITERS; ++it) {
    const int cur = it & 1;
    wait_async0();
    __syncthreads();                 // buf[cur] filled by all waves
    if (it + 1 < ITERS) {            // stage next tile into the other buffer
      const char* gnext = vtiles + (size_t)(it + 1) * TILE_BYTES + cidx;
      async_copy_b128(gnext,      &ldsb[1 - cur][cidx]);
      async_copy_b128(gnext + 16, &ldsb[1 - cur][cidx + 16]);
    }

    const int t     = s * ITERS + it;
    const int fbase = t * 32;

    // A-fragment (ISA 7.12.2, 16-bit A 16x32): lane L holds row m=L&15;
    // elems 0..7 = K = kb+{0..7}; elems 8..15 = K = 16+kb+{0..7}.
    const float* p0 = xrow + fbase + kb;
    const float* p1 = xrow + fbase + 16 + kb;
    v4f x0a = *(const v4f*)(p0);
    v4f x0b = *(const v4f*)(p0 + 4);
    v4f x1a = *(const v4f*)(p1);
    v4f x1b = *(const v4f*)(p1 + 4);
    __builtin_prefetch(p0 + 32, 0, 1);

    // preload all 8 B fragments (16x ds_load_b128) before the WMMA chain
    const v16bf* lb = (const v16bf*)&ldsb[cur][0];
    v16bf bfr[8];
#pragma unroll
    for (int nt = 0; nt < 4; ++nt) {
      bfr[nt]     = lb[nt * 32 + lane];
      bfr[4 + nt] = lb[128 + nt * 32 + lane];
    }

    float xs[16];
#pragma unroll
    for (int j = 0; j < 4; ++j) {
      xs[j]      = x0a[j];
      xs[4 + j]  = x0b[j];
      xs[8 + j]  = x1a[j];
      xs[12 + j] = x1b[j];
    }
    v16bf a, a2;
#pragma unroll
    for (int j = 0; j < 16; ++j) {
      a[j]  = (__bf16)xs[j];
      a2[j] = (__bf16)(xs[j] * xs[j]);
    }

#pragma unroll
    for (int nt = 0; nt < 4; ++nt) {
      c1[nt] = __builtin_amdgcn_wmma_f32_16x16x32_bf16(
          false, a, false, bfr[nt], (short)0, c1[nt], false, false);
      c2[nt] = __builtin_amdgcn_wmma_f32_16x16x32_bf16(
          false, a2, false, bfr[4 + nt], (short)0, c2[nt], false, false);
    }
    __syncthreads();                 // all waves done reading buf[cur]
  }

  // C/D layout: VGPR r of lane L holds element (M = r + 8*(L>>4), N = L&15).
  // c1 (xv): store per-k partials (must square after full-F sum).
  float* pc1 = c1p + (size_t)(slab * FSPLIT + s) * (16 * 64);
  const int nl = lane & 15;
#pragma unroll
  for (int nt = 0; nt < 4; ++nt) {
#pragma unroll
    for (int r = 0; r < 8; ++r) {
      pc1[(r + 8 * half) * 64 + nt * 16 + nl] = c1[nt][r];
    }
  }

  // c2 (x2v2): linear in F -> reduce over K now. Row M = r + 8*half spans
  // n = nt*16 + nl over the 16 lanes of this half: butterfly over nl bits.
#pragma unroll
  for (int r = 0; r < 8; ++r) {
    float s2 = c2[0][r] + c2[1][r] + c2[2][r] + c2[3][r];
#pragma unroll
    for (int d = 1; d < 16; d <<= 1) s2 += __shfl_xor(s2, d, 32);
    if (nl == 0) c2p[(size_t)(slab * FSPLIT + s) * 16 + (r + 8 * half)] = s2;
  }
}

// ---------------------------------------------------------------------------
// Final: per row, sum F-split xv partials, square, subtract the (already
// K-reduced) x2v2 scalars, scale by 0.5.
// ---------------------------------------------------------------------------
__global__ void fm_final(const float* __restrict__ c1p,
                         const float* __restrict__ c2p,
                         float* __restrict__ out) {
  int row = blockIdx.x * blockDim.x + threadIdx.x;
  if (row >= B_ROWS) return;
  int slab = row >> 4;
  int m    = row & 15;
  const float* base1 = c1p + (size_t)slab * FSPLIT * (16 * 64);

  float acc2 = 0.f;
#pragma unroll
  for (int s = 0; s < FSPLIT; ++s)
    acc2 += c2p[(size_t)(slab * FSPLIT + s) * 16 + m];

  float acc1 = 0.f;
  for (int k = 0; k < K_DIM; ++k) {
    float xv = 0.f;
#pragma unroll
    for (int s = 0; s < FSPLIT; ++s)
      xv += base1[s * (16 * 64) + m * 64 + k];
    acc1 += xv * xv;
  }
  out[row] = 0.5f * (acc1 - acc2);
}

// ---------------------------------------------------------------------------
extern "C" void kernel_launch(void* const* d_in, const int* in_sizes, int n_in,
                              void* d_out, int out_size, void* d_ws, size_t ws_size,
                              hipStream_t stream) {
  const float* x = (const float*)d_in[0];
  const float* v = (const float*)d_in[1];
  float* out = (float*)d_out;

  char* ws = (char*)d_ws;
  // ws layout: [vcomb bf16 1MB][c1 partials f32 8MB][c2 scalars f32 128KB]
  __bf16* vcomb = (__bf16*)ws;
  float* c1p = (float*)(ws + (size_t)FTILES * TILE_BYTES);
  float* c2p = c1p + (size_t)(B_ROWS / 16) * FSPLIT * (16 * 64);

  // prep: one thread per (ftile, ntile, lane) = 128*4*32 = 16384
  fm_prep_v<<<(FTILES * 4 * 32) / 256, 256, 0, stream>>>(v, vcomb);

  // main: 64 slab-groups x FSPLIT = 256 blocks x 256 threads (2048 waves)
  fm_main<<<(B_ROWS / (16 * WAVES_PER_BLOCK)) * FSPLIT, WAVES_PER_BLOCK * 32, 0,
            stream>>>(x, vcomb, c1p, c2p);

  // final: one thread per row
  fm_final<<<(B_ROWS + 255) / 256, 256, 0, stream>>>(c1p, c2p, out);
}
